// Decoder_44848048505602
// MI455X (gfx1250) — compile-verified, matrix operand order checked by reading, not run on hardware
//
#include <hip/hip_runtime.h>
#include <math.h>

#define B_     1024
#define T_     64
#define FEAT_  1024
#define TENC_  256
#define MAN_   128
#define HID_   2048
#define G4_    8192
#define IN_DIM_ 1408

typedef __bf16 bf16_t;
typedef bf16_t v16bf __attribute__((ext_vector_type(16)));
typedef float  v8f   __attribute__((ext_vector_type(8)));
typedef unsigned int v8u __attribute__((ext_vector_type(8)));
typedef unsigned int v4u __attribute__((ext_vector_type(4)));

__device__ __forceinline__ v8f wmma_bf16(v8u a, v8u b, v8f c) {
  return __builtin_amdgcn_wmma_f32_16x16x32_bf16(
      false, __builtin_bit_cast(v16bf, a),
      false, __builtin_bit_cast(v16bf, b),
      (short)0, c, false, false);
}

__device__ __forceinline__ unsigned short f2bf(float f) {
  unsigned u = __float_as_uint(f);
  u += 0x7FFFu + ((u >> 16) & 1u);            // round-to-nearest-even
  return (unsigned short)(u >> 16);
}
__device__ __forceinline__ float sigmoidf_(float x) { return 1.0f / (1.0f + __expf(-x)); }

// CDNA5 async global->LDS copy (ASYNCcnt-tracked, no VGPR data round-trip).
// ldsByteAddr: wave-relative LDS byte address (low 32 bits of a generic LDS pointer,
// per ISA 10.2).
__device__ __forceinline__ void async_cp16(unsigned ldsByteAddr, const void* gptr) {
  asm volatile("global_load_async_to_lds_b128 %0, %1, off"
               :: "v"(ldsByteAddr), "v"((unsigned long long)(size_t)gptr)
               : "memory");
}
__device__ __forceinline__ void wait_async0() {
  asm volatile("s_wait_asynccnt 0x0" ::: "memory");
}
__device__ __forceinline__ unsigned lds_addr_of(const void* p) {
  return (unsigned)(size_t)p;
}

// Stage a 128-row x 32-K bf16 A tile (row-major source) into an 8 KB LDS tile in
// WMMA A-fragment order, using async LDS-scatter copies: 512 x b128, 2 per thread,
// uniform control flow.  ldsTile: byte address of the tile base. src: (row 0, kBase)
// of the source; rs = row stride in ushorts.  thread -> g = tid&3 (4 K-pairs), m0 = tid>>2.
__device__ __forceinline__ void stageA(unsigned ldsTile,
                                       const unsigned short* __restrict__ src,
                                       int rs, int tid) {
  const int g = tid & 3, m0 = tid >> 2;                 // rows m0 and m0+64
  const unsigned base = ldsTile +
      (((m0 >> 4) << 8) + (((m0 & 15) | ((g & 1) << 4)) << 3) + ((g & 2) << 1)) * 4u;
  async_cp16(base,        src + m0 * rs + 8 * g);
  async_cp16(base + 4096, src + (m0 + 64) * rs + 8 * g);   // (m>>4) += 4
}

// ---------------- utility kernels ----------------

__global__ void zero_u32_kernel(unsigned* p, int n) {
  for (int i = blockIdx.x * blockDim.x + threadIdx.x; i < n; i += gridDim.x * blockDim.x)
    p[i] = 0u;
}

__global__ void f32_to_bf16_kernel(const float* __restrict__ src,
                                   unsigned short* __restrict__ dst, int n) {
  for (int i = blockIdx.x * blockDim.x + threadIdx.x; i < n; i += gridDim.x * blockDim.x)
    dst[i] = f2bf(src[i]);
}

__global__ void copy_y0_kernel(const float* __restrict__ y0, float* __restrict__ out) {
  int i = blockIdx.x * blockDim.x + threadIdx.x;         // n = B_*MAN_
  if (i < B_ * MAN_) {
    int b = i / MAN_, n = i % MAN_;
    out[b * (T_ * MAN_) + n] = y0[i];                    // t = 0 slice
  }
}

// Swizzle a (J x K) fp32 weight slab (row stride rowStride, starting col0) into
// bf16 WMMA B-fragment order: per 16x32 block, per-lane 8 contiguous dwords.
__global__ void swizzle_w_kernel(const float* __restrict__ W, int J, int rowStride,
                                 int col0, int K, unsigned short* __restrict__ dst) {
  int total = J * K;
  for (int i = blockIdx.x * blockDim.x + threadIdx.x; i < total; i += gridDim.x * blockDim.x) {
    int j = i / K, k = i % K;
    float f = W[j * rowStride + col0 + k];
    int kl = k & 31;
    int lane = (j & 15) | (((kl >> 3) & 1) << 4);
    int v = ((kl >= 16) ? 4 : 0) + ((kl & 7) >> 1);
    int blk = (j >> 4) * (K >> 5) + (k >> 5);
    dst[blk * 512 + lane * 16 + v * 2 + (k & 1)] = f2bf(f);
  }
}

// ---------------- Gconst = te @ W_te^T + b_ih + b_hh  (B x 8192 fp32) ----------------
// grid (8, 32), block 256. WG tile 128 rows x 256 cols; wave tile 64x64 (16 accums).
__global__ void __launch_bounds__(256)
gconst_kernel(const unsigned short* __restrict__ tebf, const unsigned* __restrict__ WteSwz,
              const float* __restrict__ b_ih, const float* __restrict__ b_hh,
              float* __restrict__ Gconst) {
  __shared__ __align__(32) unsigned aT[4096];            // 2 x 8 KB A tiles
  const int tid = threadIdx.x, w = tid >> 5, lane = tid & 31;
  const int rBase = blockIdx.x * 128;
  const int waveM = w & 1, waveN = w >> 1;
  const unsigned short* aSrc = tebf + rBase * TENC_;
  const unsigned ldsBase = lds_addr_of(aT);

  v8f acc[4][4];                                         // [nf][mf]
  v8f zf = {0.f, 0.f, 0.f, 0.f, 0.f, 0.f, 0.f, 0.f};
#pragma unroll
  for (int a = 0; a < 4; a++)
#pragma unroll
    for (int b = 0; b < 4; b++) acc[a][b] = zf;

  stageA(ldsBase, aSrc, TENC_, tid);
  wait_async0();
  __syncthreads();
  for (int kt = 0; kt < (TENC_ / 32); kt++) {
    unsigned* cur = aT + ((kt & 1) << 11);
    v8u bfr[4];
#pragma unroll
    for (int nf = 0; nf < 4; nf++) {
      int j0 = blockIdx.y * 256 + waveN * 64 + nf * 16;
      int blk = (j0 >> 4) * (TENC_ >> 5) + kt;
      bfr[nf] = *(const v8u*)(WteSwz + blk * 256 + lane * 8);
    }
    if (kt + 1 < (TENC_ / 32))
      stageA(ldsBase + (((kt + 1) & 1) << 13), aSrc + (kt + 1) * 32, TENC_, tid);
#pragma unroll
    for (int mf = 0; mf < 4; mf++) {
      v8u af = *(const v8u*)(cur + ((waveM * 4 + mf) << 8) + lane * 8);
#pragma unroll
      for (int nf = 0; nf < 4; nf++) acc[nf][mf] = wmma_bf16(af, bfr[nf], acc[nf][mf]);
    }
    wait_async0();
    __syncthreads();
  }
#pragma unroll
  for (int nf = 0; nf < 4; nf++) {
    int col = blockIdx.y * 256 + waveN * 64 + nf * 16 + (lane & 15);
    float bias = b_ih[col] + b_hh[col];
#pragma unroll
    for (int mf = 0; mf < 4; mf++) {
      int row0 = rBase + waveM * 64 + mf * 16 + ((lane >> 4) << 3);
#pragma unroll
      for (int r = 0; r < 8; r++)
        Gconst[(row0 + r) * G4_ + col] = acc[nf][mf][r] + bias;
    }
  }
}

// Uniform per-K-chunk segment select for A = [x_t | y | h] (boundaries are
// multiples of 32, so a 32-wide K chunk never straddles two segments).
__device__ __forceinline__ const unsigned short*
segsrc(const unsigned short* __restrict__ Xbf, const unsigned short* __restrict__ ybf,
       const unsigned short* __restrict__ hbf, int t, int kBase, int rBase, int& rs) {
  if (kBase < FEAT_)        { rs = T_ * FEAT_; return Xbf + rBase * rs + t * FEAT_ + kBase; }
  if (kBase < FEAT_ + MAN_) { rs = MAN_;       return ybf + rBase * rs + (kBase - FEAT_); }
  rs = HID_; return hbf + rBase * rs + (kBase - FEAT_ - MAN_);
}

// ---------------- fused per-step gates GEMM + LSTM cell ----------------
// gates = Gconst + [x_t | y | h] @ [Wx | Wy | Whh]^T  (K = 1024+128+2048 = 3200)
// Wave owns matching i/f/g/o columns -> cell nonlinearity fused, writes c (fp32) + h (bf16).
// grid (8, 32), block 256. WG tile: 128 rows x 64 cells; wave: 64 rows x 16 cells x 4 gates.
__global__ void __launch_bounds__(256)
lstm_step_kernel(const unsigned short* __restrict__ Xbf, const unsigned short* __restrict__ ybf,
                 const unsigned short* __restrict__ hbf_in,
                 const unsigned* __restrict__ WxSwz, const unsigned* __restrict__ WySwz,
                 const unsigned* __restrict__ WhhSwz,
                 const float* __restrict__ Gconst, float* __restrict__ c,
                 unsigned short* __restrict__ hbf_out, int t) {
  __shared__ __align__(32) unsigned aT[4096];            // 2 x 8 KB A tiles
  const int tid = threadIdx.x, w = tid >> 5, lane = tid & 31;
  const int rBase = blockIdx.x * 128;
  const int waveM = w & 1, waveC = w >> 1;
  const int cellBase = blockIdx.y * 64 + waveC * 16;
  const unsigned ldsBase = lds_addr_of(aT);

  v8f acc[4][4];                                         // [gate][mf]
#pragma unroll
  for (int g = 0; g < 4; g++) {                          // init accumulators from Gconst
    int col = g * HID_ + cellBase + (lane & 15);
#pragma unroll
    for (int mf = 0; mf < 4; mf++) {
      int row0 = rBase + waveM * 64 + mf * 16 + ((lane >> 4) << 3);
#pragma unroll
      for (int r = 0; r < 8; r++) acc[g][mf][r] = Gconst[(row0 + r) * G4_ + col];
    }
  }

  {
    int rs; const unsigned short* s = segsrc(Xbf, ybf, hbf_in, t, 0, rBase, rs);
    stageA(ldsBase, s, rs, tid);
  }
  wait_async0();
  __syncthreads();

  for (int kt = 0; kt < 100; kt++) {                     // K = 3200
    int kBase = kt * 32;
    unsigned* cur = aT + ((kt & 1) << 11);

    const unsigned* wseg; int blkK, segBlks;             // uniform (scalar) branch
    if (kBase < FEAT_)             { wseg = WxSwz;  blkK = kBase >> 5;                  segBlks = FEAT_ >> 5; }
    else if (kBase < FEAT_ + MAN_) { wseg = WySwz;  blkK = (kBase - FEAT_) >> 5;        segBlks = MAN_ >> 5; }
    else                           { wseg = WhhSwz; blkK = (kBase - FEAT_ - MAN_) >> 5; segBlks = HID_ >> 5; }

    v8u bfr[4];
#pragma unroll
    for (int g = 0; g < 4; g++) {
      int jt = (g * HID_ + cellBase) >> 4;
      bfr[g] = *(const v8u*)(wseg + (jt * segBlks + blkK) * 256 + lane * 8);
    }

    if (kt + 1 < 100) {                                  // async-stage next tile
      int rs; const unsigned short* s = segsrc(Xbf, ybf, hbf_in, t, kBase + 32, rBase, rs);
      stageA(ldsBase + (((kt + 1) & 1) << 13), s, rs, tid);
    }

#pragma unroll
    for (int mf = 0; mf < 4; mf++) {
      v8u af = *(const v8u*)(cur + ((waveM * 4 + mf) << 8) + lane * 8);
#pragma unroll
      for (int g = 0; g < 4; g++) acc[g][mf] = wmma_bf16(af, bfr[g], acc[g][mf]);
    }
    wait_async0();
    __syncthreads();
  }

  // fused LSTM cell epilogue (torch gate order i,f,g,o)
  int cell = cellBase + (lane & 15);
#pragma unroll
  for (int mf = 0; mf < 4; mf++) {
    int row0 = rBase + waveM * 64 + mf * 16 + ((lane >> 4) << 3);
#pragma unroll
    for (int r = 0; r < 8; r++) {
      int cidx = (row0 + r) * HID_ + cell;
      float iv = sigmoidf_(acc[0][mf][r]);
      float fv = sigmoidf_(acc[1][mf][r]);
      float gv = tanhf(acc[2][mf][r]);
      float ov = sigmoidf_(acc[3][mf][r]);
      float cn = fv * c[cidx] + iv * gv;
      c[cidx] = cn;
      hbf_out[cidx] = f2bf(ov * tanhf(cn));
    }
  }
}

// ---------------- y = h @ W_mu^T + b_mu -> out[:, t, :] and ybf ----------------
// grid (8, 1), block 256. WG tile 128 rows x 128 cols; wave 64x32 (8 accums).
__global__ void __launch_bounds__(256)
ymu_kernel(const unsigned short* __restrict__ hbf, const unsigned* __restrict__ WmuSwz,
           const float* __restrict__ b_mu, float* __restrict__ out,
           unsigned short* __restrict__ ybf, int t) {
  __shared__ __align__(32) unsigned aT[4096];
  const int tid = threadIdx.x, w = tid >> 5, lane = tid & 31;
  const int rBase = blockIdx.x * 128;
  const int waveM = w & 1, waveN = w >> 1;
  const unsigned short* aSrc = hbf + rBase * HID_;
  const unsigned ldsBase = lds_addr_of(aT);

  v8f acc[2][4];
  v8f zf = {0.f, 0.f, 0.f, 0.f, 0.f, 0.f, 0.f, 0.f};
#pragma unroll
  for (int a = 0; a < 2; a++)
#pragma unroll
    for (int b = 0; b < 4; b++) acc[a][b] = zf;

  stageA(ldsBase, aSrc, HID_, tid);
  wait_async0();
  __syncthreads();
  for (int kt = 0; kt < (HID_ / 32); kt++) {
    unsigned* cur = aT + ((kt & 1) << 11);
    v8u bfr[2];
#pragma unroll
    for (int nf = 0; nf < 2; nf++) {
      int jt = waveN * 2 + nf;
      bfr[nf] = *(const v8u*)(WmuSwz + (jt * (HID_ >> 5) + kt) * 256 + lane * 8);
    }
    if (kt + 1 < (HID_ / 32))
      stageA(ldsBase + (((kt + 1) & 1) << 13), aSrc + (kt + 1) * 32, HID_, tid);
#pragma unroll
    for (int mf = 0; mf < 4; mf++) {
      v8u af = *(const v8u*)(cur + ((waveM * 4 + mf) << 8) + lane * 8);
#pragma unroll
      for (int nf = 0; nf < 2; nf++) acc[nf][mf] = wmma_bf16(af, bfr[nf], acc[nf][mf]);
    }
    wait_async0();
    __syncthreads();
  }
#pragma unroll
  for (int nf = 0; nf < 2; nf++) {
    int col = waveN * 32 + nf * 16 + (lane & 15);
    float bias = b_mu[col];
#pragma unroll
    for (int mf = 0; mf < 4; mf++) {
      int row0 = rBase + waveM * 64 + mf * 16 + ((lane >> 4) << 3);
#pragma unroll
      for (int r = 0; r < 8; r++) {
        float val = acc[nf][mf][r] + bias;
        int row = row0 + r;
        out[row * (T_ * MAN_) + t * MAN_ + col] = val;
        ybf[row * MAN_ + col] = f2bf(val);
      }
    }
  }
}

// ---------------- host ----------------

extern "C" void kernel_launch(void* const* d_in, const int* in_sizes, int n_in,
                              void* d_out, int out_size, void* d_ws, size_t ws_size,
                              hipStream_t stream) {
  const float* input_repr = (const float*)d_in[0];
  const float* time_enc   = (const float*)d_in[1];
  const float* y0         = (const float*)d_in[2];
  const float* W_ih       = (const float*)d_in[3];
  const float* b_ih       = (const float*)d_in[4];
  const float* W_hh       = (const float*)d_in[5];
  const float* b_hh       = (const float*)d_in[6];
  const float* W_mu       = (const float*)d_in[7];
  const float* b_mu       = (const float*)d_in[8];
  float* out = (float*)d_out;

  char* ws = (char*)d_ws;
  size_t off = 0;
  auto alloc = [&](size_t bytes) -> char* {
    char* p = ws + off;
    off = (off + bytes + 255) & ~(size_t)255;
    return p;
  };
  unsigned*       WxSwz  = (unsigned*)alloc((size_t)G4_ * FEAT_ * 2);
  unsigned*       WySwz  = (unsigned*)alloc((size_t)G4_ * MAN_ * 2);
  unsigned*       WteSwz = (unsigned*)alloc((size_t)G4_ * TENC_ * 2);
  unsigned*       WhhSwz = (unsigned*)alloc((size_t)G4_ * HID_ * 2);
  unsigned*       WmuSwz = (unsigned*)alloc((size_t)MAN_ * HID_ * 2);
  float*          Gconst = (float*)alloc((size_t)B_ * G4_ * 4);
  unsigned short* Xbf    = (unsigned short*)alloc((size_t)B_ * T_ * FEAT_ * 2);
  unsigned short* tebf   = (unsigned short*)alloc((size_t)B_ * TENC_ * 2);
  unsigned short* ybf    = (unsigned short*)alloc((size_t)B_ * MAN_ * 2);
  unsigned short* h0bf   = (unsigned short*)alloc((size_t)B_ * HID_ * 2);
  unsigned short* h1bf   = (unsigned short*)alloc((size_t)B_ * HID_ * 2);
  float*          cbuf   = (float*)alloc((size_t)B_ * HID_ * 4);
  (void)ws_size; (void)in_sizes; (void)n_in; (void)out_size;

  auto gsz = [](int n) { int g = (n + 255) / 256; return g > 8192 ? 8192 : g; };

  // One-time (per call) weight swizzles to bf16 fragment order
  swizzle_w_kernel<<<gsz(G4_ * FEAT_), 256, 0, stream>>>(W_ih, G4_, IN_DIM_, 0, FEAT_, (unsigned short*)WxSwz);
  swizzle_w_kernel<<<gsz(G4_ * MAN_),  256, 0, stream>>>(W_ih, G4_, IN_DIM_, FEAT_, MAN_, (unsigned short*)WySwz);
  swizzle_w_kernel<<<gsz(G4_ * TENC_), 256, 0, stream>>>(W_ih, G4_, IN_DIM_, FEAT_ + MAN_, TENC_, (unsigned short*)WteSwz);
  swizzle_w_kernel<<<gsz(G4_ * HID_),  256, 0, stream>>>(W_hh, G4_, HID_, 0, HID_, (unsigned short*)WhhSwz);
  swizzle_w_kernel<<<gsz(MAN_ * HID_), 256, 0, stream>>>(W_mu, MAN_, HID_, 0, HID_, (unsigned short*)WmuSwz);

  // Activations to bf16; state init
  f32_to_bf16_kernel<<<gsz(B_ * T_ * FEAT_), 256, 0, stream>>>(input_repr, Xbf, B_ * T_ * FEAT_);
  f32_to_bf16_kernel<<<gsz(B_ * TENC_), 256, 0, stream>>>(time_enc, tebf, B_ * TENC_);
  f32_to_bf16_kernel<<<gsz(B_ * MAN_), 256, 0, stream>>>(y0, ybf, B_ * MAN_);
  zero_u32_kernel<<<gsz(B_ * HID_), 256, 0, stream>>>((unsigned*)cbuf, B_ * HID_);
  zero_u32_kernel<<<gsz(B_ * HID_ / 2), 256, 0, stream>>>((unsigned*)h0bf, B_ * HID_ / 2);
  copy_y0_kernel<<<gsz(B_ * MAN_), 256, 0, stream>>>(y0, out);

  // Time-invariant gate pre-accumulation
  gconst_kernel<<<dim3(8, 32), 256, 0, stream>>>(tebf, WteSwz, b_ih, b_hh, Gconst);

  // Serial recurrence: 63 steps, h ping-pong across launches
  for (int t = 1; t < T_; t++) {
    unsigned short* hin  = (t & 1) ? h0bf : h1bf;
    unsigned short* hout = (t & 1) ? h1bf : h0bf;
    lstm_step_kernel<<<dim3(8, 32), 256, 0, stream>>>(Xbf, ybf, hin, WxSwz, WySwz, WhhSwz,
                                                      Gconst, cbuf, hout, t);
    ymu_kernel<<<dim3(8, 1), 256, 0, stream>>>(hout, WmuSwz, b_mu, out, ybf, t);
  }
}
